// SelfAttention_34248069218852
// MI455X (gfx1250) — matrix operand order repaired and checked by reading, not compiled, not run on hardware
//
#include <hip/hip_runtime.h>

#define B_   4
#define S_   2048
#define D_   1024
#define H_   16
#define DH_  64

typedef __attribute__((ext_vector_type(16))) __bf16 bf16x16;
typedef __attribute__((ext_vector_type(8)))  __bf16 bf16x8;
typedef __attribute__((ext_vector_type(8)))  float  f32x8;

union BFrag {
  bf16x16 v;
  bf16x8  h[2];
};

union V4B8 {
  uint4   u;
  __bf16  e[8];
};

static __device__ __forceinline__ f32x8 wmma_bf16(const bf16x16& a, const bf16x16& b, f32x8 c) {
  // D = A(16x32 bf16) * B(32x16 bf16) + C(16x16 f32)
  return __builtin_amdgcn_wmma_f32_16x16x32_bf16(false, a, false, b, (short)0, c, false, false);
}

// LDS byte offset of a __shared__ object = low 32 bits of its generic address
// (LDS aperture, ISA sec. 10.2).
static __device__ __forceinline__ unsigned lds_off_of(const void* p) {
  return (unsigned)(uintptr_t)p;
}

// CDNA5 async copy: 16 bytes global -> LDS per lane, tracked by ASYNCcnt.
static __device__ __forceinline__ void async_copy_b128(unsigned lds_off, const void* gaddr) {
  asm volatile("global_load_async_to_lds_b128 %0, %1, off"
               :: "v"(lds_off), "v"(gaddr) : "memory");
}
static __device__ __forceinline__ void wait_async0() {
  asm volatile("s_wait_asynccnt 0" ::: "memory");
}

// reductions across the 16-lane half (rows of a 16x16 C tile live per half in wave32)
static __device__ __forceinline__ float half16_max(float v) {
  #pragma unroll
  for (int off = 1; off < 16; off <<= 1) v = fmaxf(v, __shfl_xor(v, off, 32));
  return v;
}
static __device__ __forceinline__ float half16_sum(float v) {
  #pragma unroll
  for (int off = 1; off < 16; off <<= 1) v += __shfl_xor(v, off, 32);
  return v;
}

// ---------------------------------------------------------------------------
// One-time converts (off the critical path; x and W live in L2 afterwards).
// ---------------------------------------------------------------------------
__global__ __launch_bounds__(256)
void cvt_f32_to_bf16(const float* __restrict__ in, __bf16* __restrict__ out) {
  size_t i = ((size_t)blockIdx.x * 256 + threadIdx.x) * 8;
  float4 a = *(const float4*)(in + i);
  float4 b = *(const float4*)(in + i + 4);
  V4B8 t;
  t.e[0] = (__bf16)a.x; t.e[1] = (__bf16)a.y; t.e[2] = (__bf16)a.z; t.e[3] = (__bf16)a.w;
  t.e[4] = (__bf16)b.x; t.e[5] = (__bf16)b.y; t.e[6] = (__bf16)b.z; t.e[7] = (__bf16)b.w;
  *(uint4*)(out + i) = t.u;
}

// W[k][n] fp32 -> Wt[n][k] bf16 (so GEMM B-tiles are straight async copies)
__global__ __launch_bounds__(256)
void cvt_w_transpose(const float* __restrict__ W, __bf16* __restrict__ Wt) {
  int idx = blockIdx.x * 256 + threadIdx.x;   // 0 .. 1024*1024/4 - 1
  int k   = idx >> 8;                         // 0..1023
  int n0  = (idx & 255) * 4;
  float4 f = *(const float4*)(W + (size_t)k * D_ + n0);
  Wt[(size_t)(n0 + 0) * D_ + k] = (__bf16)f.x;
  Wt[(size_t)(n0 + 1) * D_ + k] = (__bf16)f.y;
  Wt[(size_t)(n0 + 2) * D_ + k] = (__bf16)f.z;
  Wt[(size_t)(n0 + 3) * D_ + k] = (__bf16)f.w;
}

// ---------------------------------------------------------------------------
// Fully-async double-buffered GEMM: out = A(8192x1024) * W + bias
// A: bf16 row-major [m][k]; Wt: bf16 transposed [n][k].
// Both tiles stream via GLOBAL_LOAD_ASYNC_TO_LDS_B128; tile k+1's DMA runs
// under tile k's WMMAs; one s_wait_asynccnt + one barrier per k-step.
// head_layout=1: write bf16 into (B,H,S,DH); else fp32 row-major.
// Block: 256 thr (8 waves), tile 128x128x32; wave sub-tile 32x64 (2x4 WMMA).
// ---------------------------------------------------------------------------
__global__ __launch_bounds__(256)
void gemm_bf16_wmma(const __bf16* __restrict__ A, const __bf16* __restrict__ Wt,
                    const float* __restrict__ bias,
                    __bf16* __restrict__ out_bf, float* __restrict__ out_f,
                    int head_layout)
{
  constexpr int MTILE = 128, NTILE = 128, KTILE = 32, LDT = 40; // padded stride
  constexpr int Kdim = D_, Ndim = D_;
  __shared__ __bf16 lds_a[2][MTILE * LDT];   // [m][k], k contiguous
  __shared__ __bf16 lds_w[2][NTILE * LDT];   // [n][k], k contiguous

  const int tid  = threadIdx.x;
  const int wave = tid >> 5;
  const int lane = tid & 31;
  const int hi   = lane >> 4;   // lane half selects K/M sub-range per ISA layout
  const int ln   = lane & 15;
  const int wm   = wave & 3;    // 4 waves along M
  const int wn   = wave >> 2;   // 2 waves along N
  const int m0   = blockIdx.x * MTILE;
  const int n0   = blockIdx.y * NTILE;

  // per-thread async-copy slots (A and W tiles share the 128x32 shape)
  const int crow = tid >> 2;           // 0..63 (i=0) / +64 (i=1)
  const int ckq  = (tid & 3) * 8;      // 0,8,16,24

  auto issue_tile = [&](int buf, int k0) {
    #pragma unroll
    for (int i = 0; i < 2; ++i) {
      int row = crow + i * 64;
      async_copy_b128(lds_off_of(&lds_a[buf][row * LDT + ckq]),
                      A  + (size_t)(m0 + row) * Kdim + k0 + ckq);
      async_copy_b128(lds_off_of(&lds_w[buf][row * LDT + ckq]),
                      Wt + (size_t)(n0 + row) * Kdim + k0 + ckq);
    }
  };

  f32x8 acc[2][4] = {};
  issue_tile(0, 0);

  for (int kt = 0; kt < Kdim / KTILE; ++kt) {
    const int k0  = kt * KTILE;
    const int cur = kt & 1;

    wait_async0();       // this wave's tile kt landed
    __syncthreads();     // everyone's tile kt visible; buffer cur^1 free

    if (k0 + KTILE < Kdim) {
      issue_tile(cur ^ 1, k0 + KTILE);
      // warm L2 two tiles ahead under the async engine
      if (k0 + 2 * KTILE < Kdim) {
        __builtin_prefetch(A  + (size_t)(m0 + crow) * Kdim + k0 + 2 * KTILE + ckq, 0, 1);
        __builtin_prefetch(Wt + (size_t)(n0 + crow) * Kdim + k0 + 2 * KTILE + ckq, 0, 1);
      }
    }

    // A fragment: row = lane, K elems {base..base+7, base+16..base+23}, base=hi*8
    BFrag afrag[2];
    #pragma unroll
    for (int fm = 0; fm < 2; ++fm) {
      const __bf16* p = &lds_a[cur][(wm * 32 + fm * 16 + ln) * LDT + hi * 8];
      afrag[fm].h[0] = *(const bf16x8*)p;
      afrag[fm].h[1] = *(const bf16x8*)(p + 16);
    }
    // B fragment: col = lane, K elems {base..base+7, base+16..base+23}, base=hi*8
    #pragma unroll
    for (int fn = 0; fn < 4; ++fn) {
      BFrag bfrag;
      const __bf16* p = &lds_w[cur][(wn * 64 + fn * 16 + ln) * LDT + hi * 8];
      bfrag.h[0] = *(const bf16x8*)p;
      bfrag.h[1] = *(const bf16x8*)(p + 16);
      #pragma unroll
      for (int fm = 0; fm < 2; ++fm)
        acc[fm][fn] = wmma_bf16(afrag[fm].v, bfrag.v, acc[fm][fn]);
    }
  }

  // Epilogue: C layout => M = r + hi*8, N = ln (per 16x16 tile)
  #pragma unroll
  for (int fm = 0; fm < 2; ++fm) {
    #pragma unroll
    for (int fn = 0; fn < 4; ++fn) {
      int n = n0 + wn * 64 + fn * 16 + ln;
      float bv = bias[n];
      #pragma unroll
      for (int r = 0; r < 8; ++r) {
        int m = m0 + wm * 32 + fm * 16 + hi * 8 + r;
        float val = acc[fm][fn][r] + bv;
        if (head_layout) {
          int bb = m >> 11;          // / S_
          int ss = m & (S_ - 1);
          int hh = n >> 6;           // / DH_
          int dd = n & (DH_ - 1);
          out_bf[(((size_t)(bb * H_ + hh) * S_) + ss) * DH_ + dd] = (__bf16)val;
        } else {
          out_f[(size_t)m * Ndim + n] = val;
        }
      }
    }
  }
}

// ---------------------------------------------------------------------------
// Flash attention: per block 128 query rows of one (b,h); stream 64-key tiles.
// 8 waves; wave owns 16 query rows. Online softmax in fp32, WMMA bf16 for
// S = Q*K^T and O += P*V.
// Pipelining: Q + K tiles via GLOBAL_LOAD_ASYNC_TO_LDS_B128 (double-buffered
// K/V); next V tile's global loads are issued before compute and its
// transposed LDS stores deferred until after compute.
// ---------------------------------------------------------------------------
__global__ __launch_bounds__(256)
void flash_attn_wmma(const __bf16* __restrict__ Q, const __bf16* __restrict__ K,
                     const __bf16* __restrict__ V,
                     const unsigned char* __restrict__ amask,
                     const int* __restrict__ causal_flag,
                     __bf16* __restrict__ attn_out)
{
  constexpr int BR = 128, BC = 64, LDT = 72;     // padded stride (16B-aligned rows)
  __shared__ __bf16 lds_q[BR * LDT];             // [qrow][dh]
  __shared__ __bf16 lds_k[2][BC * LDT];          // [key][dh]  (B-frag of QK^T: K=dh)
  __shared__ __bf16 lds_v[2][DH_ * LDT];         // [dh][key]  (B-frag of PV:  K=key)
  __shared__ __bf16 lds_p[8 * 16 * LDT];         // per-wave P staging (16 x 64)

  const int tid  = threadIdx.x;
  const int wave = tid >> 5;
  const int lane = tid & 31;
  const int hi   = lane >> 4;
  const int ln   = lane & 15;
  const int q0   = blockIdx.x * BR;
  const int bh   = blockIdx.y;
  const int b    = bh >> 4;     // / H_
  const int h    = bh & 15;
  const int causal = *causal_flag;

  const size_t base = (size_t)bh * S_ * DH_;

  // ---- Q tile: async copy, 16KB ----
  #pragma unroll
  for (int i = 0; i < 4; ++i) {
    int idx = tid + i * 256;
    int row = idx >> 3;
    int c   = (idx & 7) * 8;
    async_copy_b128(lds_off_of(&lds_q[row * LDT + c]),
                    Q + base + (size_t)(q0 + row) * DH_ + c);
  }

  // per-thread V staging indices (same every tile)
  const int vkey0 = tid >> 3;            // 0..31  (i=0) / +32 (i=1)
  const int vd0   = (tid & 7) * 8;

  // ---- prologue: tile 0 ----
  #pragma unroll
  for (int i = 0; i < 2; ++i) {
    int idx = tid + i * 256;
    int row = idx >> 3;
    int c   = (idx & 7) * 8;
    async_copy_b128(lds_off_of(&lds_k[0][row * LDT + c]),
                    K + base + (size_t)row * DH_ + c);
  }
  {
    V4B8 t0, t1;
    t0.u = *(const uint4*)(V + base + (size_t)vkey0 * DH_ + vd0);
    t1.u = *(const uint4*)(V + base + (size_t)(vkey0 + 32) * DH_ + vd0);
    #pragma unroll
    for (int j = 0; j < 8; ++j) {
      lds_v[0][(vd0 + j) * LDT + vkey0]      = t0.e[j];
      lds_v[0][(vd0 + j) * LDT + vkey0 + 32] = t1.e[j];
    }
  }

  f32x8 o[4] = {};
  float mrow[8], lrow[8];
  #pragma unroll
  for (int r = 0; r < 8; ++r) { mrow[r] = -3.0e38f; lrow[r] = 0.f; }

  const int ktend = causal ? (q0 / BC + 2) : (S_ / BC);

  for (int kt = 0; kt < ktend; ++kt) {
    const int c0  = kt * BC;
    const int cur = kt & 1;
    const int nxt = cur ^ 1;
    const bool havenext = (kt + 1) < ktend;

    wait_async0();       // K (and Q on first iter) landed in LDS
    __syncthreads();     // all waves' tiles visible; prev buffers free

    // ---- issue next tile's loads before computing on this one ----
    V4B8 t0, t1;
    if (havenext) {
      const int c0n = c0 + BC;
      #pragma unroll
      for (int i = 0; i < 2; ++i) {
        int idx = tid + i * 256;
        int row = idx >> 3;
        int c   = (idx & 7) * 8;
        async_copy_b128(lds_off_of(&lds_k[nxt][row * LDT + c]),
                        K + base + (size_t)(c0n + row) * DH_ + c);
      }
      t0.u = *(const uint4*)(V + base + (size_t)(c0n + vkey0) * DH_ + vd0);
      t1.u = *(const uint4*)(V + base + (size_t)(c0n + vkey0 + 32) * DH_ + vd0);
    }

    // ---- S = Q K^T  (16 x 64 per wave) ----
    f32x8 sacc[4] = {};
    #pragma unroll
    for (int ks = 0; ks < 2; ++ks) {
      BFrag aq;
      const __bf16* ap = &lds_q[(wave * 16 + ln) * LDT + ks * 32 + hi * 8];
      aq.h[0] = *(const bf16x8*)ap;
      aq.h[1] = *(const bf16x8*)(ap + 16);
      #pragma unroll
      for (int fn = 0; fn < 4; ++fn) {
        BFrag bk;
        const __bf16* bp = &lds_k[cur][(fn * 16 + ln) * LDT + ks * 32 + hi * 8];
        bk.h[0] = *(const bf16x8*)bp;
        bk.h[1] = *(const bf16x8*)(bp + 16);
        sacc[fn] = wmma_bf16(aq.v, bk.v, sacc[fn]);
      }
    }

    // ---- scale, mask, online softmax ----
    bool mk[4]; int ncol[4];
    #pragma unroll
    for (int fn = 0; fn < 4; ++fn) {
      ncol[fn] = c0 + fn * 16 + ln;
      mk[fn]   = amask[(size_t)b * S_ + ncol[fn]] != 0;
    }
    float al[8];
    #pragma unroll
    for (int r = 0; r < 8; ++r) {
      const int qrow = q0 + wave * 16 + hi * 8 + r;
      float mx = -3.0e38f;
      #pragma unroll
      for (int fn = 0; fn < 4; ++fn) {
        float v = sacc[fn][r] * 0.125f;  // 1/sqrt(64)
        if ((causal && ncol[fn] > qrow) || !mk[fn]) v = -3.0e38f;
        sacc[fn][r] = v;
        mx = fmaxf(mx, v);
      }
      mx = half16_max(mx);
      const float mn = fmaxf(mrow[r], mx);
      al[r] = __expf(mrow[r] - mn);
      float rs = 0.f;
      #pragma unroll
      for (int fn = 0; fn < 4; ++fn) {
        float v = sacc[fn][r];
        float p = (v <= -1.0e37f) ? 0.f : __expf(v - mn);
        sacc[fn][r] = p;
        rs += p;
      }
      rs = half16_sum(rs);
      lrow[r] = lrow[r] * al[r] + rs;
      mrow[r] = mn;
    }
    #pragma unroll
    for (int fd = 0; fd < 4; ++fd)
      #pragma unroll
      for (int r = 0; r < 8; ++r)
        o[fd][r] *= al[r];

    // ---- relayout P through per-wave LDS, then O += P V ----
    __bf16* pw = &lds_p[wave * 16 * LDT];
    #pragma unroll
    for (int fn = 0; fn < 4; ++fn)
      #pragma unroll
      for (int r = 0; r < 8; ++r)
        pw[(hi * 8 + r) * LDT + fn * 16 + ln] = (__bf16)sacc[fn][r];

    #pragma unroll
    for (int ks = 0; ks < 2; ++ks) {
      BFrag ap;
      const __bf16* pp = &lds_p[(wave * 16 + ln) * LDT + ks * 32 + hi * 8];
      ap.h[0] = *(const bf16x8*)pp;
      ap.h[1] = *(const bf16x8*)(pp + 16);
      #pragma unroll
      for (int fd = 0; fd < 4; ++fd) {
        BFrag bv;
        const __bf16* vp = &lds_v[cur][(fd * 16 + ln) * LDT + ks * 32 + hi * 8];
        bv.h[0] = *(const bf16x8*)vp;
        bv.h[1] = *(const bf16x8*)(vp + 16);
        o[fd] = wmma_bf16(ap.v, bv.v, o[fd]);
      }
    }

    // ---- deferred transposed store of next V tile (loads hid under compute) ----
    if (havenext) {
      #pragma unroll
      for (int j = 0; j < 8; ++j) {
        lds_v[nxt][(vd0 + j) * LDT + vkey0]      = t0.e[j];
        lds_v[nxt][(vd0 + j) * LDT + vkey0 + 32] = t1.e[j];
      }
    }
  }

  // ---- normalize and write attn (B,S,D) bf16 ----
  #pragma unroll
  for (int r = 0; r < 8; ++r) {
    float inv = (lrow[r] > 0.f) ? (1.0f / lrow[r]) : 0.f;
    const int srow = q0 + wave * 16 + hi * 8 + r;
    #pragma unroll
    for (int fd = 0; fd < 4; ++fd) {
      int d = h * DH_ + fd * 16 + ln;
      attn_out[((size_t)b * S_ + srow) * D_ + d] = (__bf16)(o[fd][r] * inv);
    }
  }
}

// ---------------------------------------------------------------------------
extern "C" void kernel_launch(void* const* d_in, const int* in_sizes, int n_in,
                              void* d_out, int out_size, void* d_ws, size_t ws_size,
                              hipStream_t stream) {
  (void)in_sizes; (void)n_in; (void)out_size; (void)ws_size;
  const float*         x      = (const float*)d_in[0];
  const unsigned char* amask  = (const unsigned char*)d_in[1];
  const int*           causal = (const int*)d_in[2];
  const float* Wq = (const float*)d_in[3];
  const float* bq = (const float*)d_in[4];
  const float* Wk = (const float*)d_in[5];
  const float* bk = (const float*)d_in[6];
  const float* Wv = (const float*)d_in[7];
  const float* bv = (const float*)d_in[8];
  const float* Wo = (const float*)d_in[9];
  const float* bo = (const float*)d_in[10];
  float* out = (float*)d_out;

  const size_t qkvElems = (size_t)B_ * H_ * S_ * DH_;   // 8,388,608 bf16
  const size_t wElems   = (size_t)D_ * D_;              // 1,048,576 bf16
  __bf16* qb   = (__bf16*)d_ws;
  __bf16* kb   = qb + qkvElems;
  __bf16* vb   = kb + qkvElems;
  __bf16* attn = vb + qkvElems;
  __bf16* xb   = attn + qkvElems;       // x in bf16
  __bf16* wqT  = xb + qkvElems;         // transposed bf16 weights
  __bf16* wkT  = wqT + wElems;
  __bf16* wvT  = wkT + wElems;
  __bf16* woT  = wvT + wElems;

  dim3 gthr(256);
  // one-time precision/layout conversion (inputs then live in L2 as bf16)
  hipLaunchKernelGGL(cvt_f32_to_bf16, dim3((unsigned)(qkvElems / (256 * 8))), gthr, 0, stream, x, xb);
  hipLaunchKernelGGL(cvt_w_transpose, dim3((unsigned)(wElems / (256 * 4))), gthr, 0, stream, Wq, wqT);
  hipLaunchKernelGGL(cvt_w_transpose, dim3((unsigned)(wElems / (256 * 4))), gthr, 0, stream, Wk, wkT);
  hipLaunchKernelGGL(cvt_w_transpose, dim3((unsigned)(wElems / (256 * 4))), gthr, 0, stream, Wv, wvT);
  hipLaunchKernelGGL(cvt_w_transpose, dim3((unsigned)(wElems / (256 * 4))), gthr, 0, stream, Wo, woT);

  dim3 gblk(B_ * S_ / 128, D_ / 128);   // (64, 8)
  hipLaunchKernelGGL(gemm_bf16_wmma, gblk, gthr, 0, stream, xb, wqT, bq, qb, (float*)nullptr, 1);
  hipLaunchKernelGGL(gemm_bf16_wmma, gblk, gthr, 0, stream, xb, wkT, bk, kb, (float*)nullptr, 1);
  hipLaunchKernelGGL(gemm_bf16_wmma, gblk, gthr, 0, stream, xb, wvT, bv, vb, (float*)nullptr, 1);
  hipLaunchKernelGGL(flash_attn_wmma, dim3(S_ / 128, B_ * H_), gthr, 0, stream,
                     qb, kb, vb, amask, causal, attn);
  hipLaunchKernelGGL(gemm_bf16_wmma, gblk, gthr, 0, stream, attn, woT, bo, (__bf16*)nullptr, out, 0);
}
